// GraphSAGE_Net_66614942761183
// MI455X (gfx1250) — matrix operand order repaired and checked by reading, not compiled, not run on hardware
//
#include <hip/hip_runtime.h>

// ---------------------------------------------------------------------------
// GraphSAGE 2-layer (SAGEConv mean aggr) for MI455X / gfx1250.
//   layer: out = relu?( mean_{j->i} x_j @ W_l + b + x_i @ W_r )
// Scatter phase is bandwidth/atomic bound (~2 GB moved); GEMM phase uses
// v_wmma_f32_16x16x32_f16 (f16 in, f32 accumulate). Weights are staged into
// LDS pre-swizzled into the WMMA B-fragment layout so each fragment is one
// 32-byte LDS vector load instead of 16 ds_load_u16.
// ---------------------------------------------------------------------------

typedef __attribute__((ext_vector_type(16))) _Float16 v16h;
typedef __attribute__((ext_vector_type(8)))  float    v8f;

#define SAGE_NODES 100000
#define SAGE_EDGES 1600000
#define SAGE_K     64        // feature dim entering each layer

// --------------------------------------------------------------------------
__global__ void sage_zero_f32(float* __restrict__ p, int n) {
    int i = blockIdx.x * blockDim.x + threadIdx.x;
    if (i < n) p[i] = 0.0f;
}

// One thread per (edge, channel). c==0 thread also bumps the degree.
__global__ void sage_scatter(const float* __restrict__ feat,
                             const int*   __restrict__ src,
                             const int*   __restrict__ dst,
                             float* __restrict__ msg,
                             float* __restrict__ deg) {
    long long t = (long long)blockIdx.x * blockDim.x + threadIdx.x;
    if (t >= (long long)SAGE_EDGES * SAGE_K) return;
    int e = (int)(t >> 6);          // SAGE_K == 64
    int c = (int)(t & 63);
    int s = src[e];
    int d = dst[e];
    atomicAdd(&msg[(long long)d * SAGE_K + c], feat[(long long)s * SAGE_K + c]);
    if (deg != nullptr && c == 0) atomicAdd(&deg[d], 1.0f);
}

// --------------------------------------------------------------------------
// Fused SAGE layer GEMM: out[M x N] = act( (msg/deg) @ Wl + bias + x @ Wr )
// One wave per 16x16 output tile; 8 waves (256 threads) per block.
// K = 64 -> 2 k-steps of 32 per operand matrix -> 4 WMMAs per tile.
//
// LDS weight layout (per matrix): fragment-major. For col-tile tn and lane l,
// the 32 halfs that lane needs live contiguously at ((tn*32 + l)*32):
//   [kstep*16 + j] = W[kstep*32 + (j>>3)*16 + (l>>4)*8 + (j&7)][tn*16 + (l&15)]   (f16)
// matching the A fragment K order: lanes 0-15 hold K rows k0..k0+7, k0+16..23,
// lanes 16-31 hold K rows k0+8..15, k0+24..31, N = lane&15.
template <int N, bool RELU>
__global__ __launch_bounds__(256)
void sage_layer_wmma(const float* __restrict__ msg,
                     const float* __restrict__ deg,
                     const float* __restrict__ xin,
                     const float* __restrict__ Wl,
                     const float* __restrict__ bias,
                     const float* __restrict__ Wr,
                     float* __restrict__ out) {
    constexpr int K = SAGE_K;                 // 64
    constexpr int tilesN = N / 16;
    constexpr int FRAG_ELEMS = tilesN * 32 * 32;   // == K * N

    __shared__ __align__(64) _Float16 sWl[FRAG_ELEMS];
    __shared__ __align__(64) _Float16 sWr[FRAG_ELEMS];

    // Stage weights into LDS, f16-converted and pre-swizzled to fragment layout.
    for (int d = threadIdx.x; d < FRAG_ELEMS; d += 256) {
        const int tn    = d >> 10;            // / (32*32)
        const int rem   = d & 1023;
        const int l     = rem >> 5;           // lane this element belongs to
        const int e     = rem & 31;           // element within lane's fragment
        const int kstep = e >> 4;
        const int j     = e & 15;
        const int k     = kstep * 32 + (j >> 3) * 16 + (l >> 4) * 8 + (j & 7);
        const int n     = tn * 16 + (l & 15);
        sWl[d] = (_Float16)Wl[k * N + n];
        sWr[d] = (_Float16)Wr[k * N + n];
    }
    __syncthreads();

    const int lane = threadIdx.x & 31;
    const int wave = threadIdx.x >> 5;
    const long long numTiles = (long long)(SAGE_NODES / 16) * tilesN;
    const long long tile = (long long)blockIdx.x * 8 + wave;
    if (tile >= numTiles) return;             // wave-uniform: EXEC stays all-1s

    const int m0   = (int)(tile / tilesN) * 16;
    const int tn   = (int)(tile % tilesN);
    const int n0   = tn * 16;
    const int l15  = lane & 15;
    const int half = lane >> 4;               // 0: lanes 0-15, 1: lanes 16-31
    const int mrow = m0 + l15;                // A-fragment row for this lane
    const int ncol = n0 + l15;                // B/C-fragment column

    const float rdeg = 1.0f / fmaxf(deg[mrow], 1.0f);
    const float4* __restrict__ mr4 =
        (const float4*)(msg + (long long)mrow * K);
    const float4* __restrict__ xr4 =
        (const float4*)(xin + (long long)mrow * K);
    const _Float16* __restrict__ fragL = &sWl[(tn * 32 + lane) * 32];
    const _Float16* __restrict__ fragR = &sWr[(tn * 32 + lane) * 32];

    v8f c = {};
#pragma unroll
    for (int kstep = 0; kstep < 2; ++kstep) {
        // ---- A fragments (16-bit A 16x32 layout, ISA 7.12.2) ----
        // lanes 0-15: elems 0..7 -> K=k0+0..7,  elems 8..15 -> K=k0+16..23
        // lanes16-31: elems 0..7 -> K=k0+8..15, elems 8..15 -> K=k0+24..31
        const int q = kstep * 8 + half * 2;   // float4 index of K=ka..ka+3
        const float4 m0v = mr4[q],     m1v = mr4[q + 1];
        const float4 m2v = mr4[q + 4], m3v = mr4[q + 5];
        const float4 x0v = xr4[q],     x1v = xr4[q + 1];
        const float4 x2v = xr4[q + 4], x3v = xr4[q + 5];
        v16h am, ax;
        am[0]  = (_Float16)(m0v.x * rdeg); am[1]  = (_Float16)(m0v.y * rdeg);
        am[2]  = (_Float16)(m0v.z * rdeg); am[3]  = (_Float16)(m0v.w * rdeg);
        am[4]  = (_Float16)(m1v.x * rdeg); am[5]  = (_Float16)(m1v.y * rdeg);
        am[6]  = (_Float16)(m1v.z * rdeg); am[7]  = (_Float16)(m1v.w * rdeg);
        am[8]  = (_Float16)(m2v.x * rdeg); am[9]  = (_Float16)(m2v.y * rdeg);
        am[10] = (_Float16)(m2v.z * rdeg); am[11] = (_Float16)(m2v.w * rdeg);
        am[12] = (_Float16)(m3v.x * rdeg); am[13] = (_Float16)(m3v.y * rdeg);
        am[14] = (_Float16)(m3v.z * rdeg); am[15] = (_Float16)(m3v.w * rdeg);
        ax[0]  = (_Float16)x0v.x; ax[1]  = (_Float16)x0v.y;
        ax[2]  = (_Float16)x0v.z; ax[3]  = (_Float16)x0v.w;
        ax[4]  = (_Float16)x1v.x; ax[5]  = (_Float16)x1v.y;
        ax[6]  = (_Float16)x1v.z; ax[7]  = (_Float16)x1v.w;
        ax[8]  = (_Float16)x2v.x; ax[9]  = (_Float16)x2v.y;
        ax[10] = (_Float16)x2v.z; ax[11] = (_Float16)x2v.w;
        ax[12] = (_Float16)x3v.x; ax[13] = (_Float16)x3v.y;
        ax[14] = (_Float16)x3v.z; ax[15] = (_Float16)x3v.w;

        // ---- B fragments: one 32-byte vector load each from LDS ----
        const v16h bl = *(const v16h*)(fragL + kstep * 16);
        const v16h br = *(const v16h*)(fragR + kstep * 16);

        c = __builtin_amdgcn_wmma_f32_16x16x32_f16(false, am, false, bl,
                                                   (short)0, c, false, false);
        c = __builtin_amdgcn_wmma_f32_16x16x32_f16(false, ax, false, br,
                                                   (short)0, c, false, false);
    }

    // C/D layout: VGPR r -> M = m0 + r + half*8, N = ncol
    const float bn = bias[ncol];
#pragma unroll
    for (int r = 0; r < 8; ++r) {
        float v = c[r] + bn;
        if (RELU) v = fmaxf(v, 0.0f);
        out[(long long)(m0 + r + half * 8) * N + ncol] = v;
    }
}

// --------------------------------------------------------------------------
extern "C" void kernel_launch(void* const* d_in, const int* in_sizes, int n_in,
                              void* d_out, int out_size, void* d_ws, size_t ws_size,
                              hipStream_t stream) {
    (void)in_sizes; (void)n_in; (void)out_size; (void)ws_size;

    const float* x   = (const float*)d_in[0];
    const int*   ei  = (const int*)  d_in[1];   // (2, E): row0=src, row1=dst
    const float* W1l = (const float*)d_in[2];
    const float* b1  = (const float*)d_in[3];
    const float* W1r = (const float*)d_in[4];
    const float* W2l = (const float*)d_in[5];
    const float* b2  = (const float*)d_in[6];
    const float* W2r = (const float*)d_in[7];
    float* out = (float*)d_out;

    const int* src = ei;
    const int* dst = ei + SAGE_EDGES;

    float* msg = (float*)d_ws;                          // 100000*64 f32
    float* deg = msg + (size_t)SAGE_NODES * SAGE_K;     // 100000 f32
    float* h   = deg + SAGE_NODES;                      // 100000*64 f32

    const int nzero1 = SAGE_NODES * SAGE_K + SAGE_NODES;   // msg + deg
    const int nzero2 = SAGE_NODES * SAGE_K;                // msg only
    const long long scatThreads = (long long)SAGE_EDGES * SAGE_K;
    const int scatBlocks = (int)((scatThreads + 255) / 256);
    const int gemmBlocks1 = ((SAGE_NODES / 16) * (SAGE_K / 16) + 7) / 8;  // N=64
    const int gemmBlocks2 = ((SAGE_NODES / 16) * 1 + 7) / 8;              // N=16

    // ---- layer 1 ----
    sage_zero_f32<<<(nzero1 + 255) / 256, 256, 0, stream>>>(msg, nzero1);
    sage_scatter<<<scatBlocks, 256, 0, stream>>>(x, src, dst, msg, deg);
    sage_layer_wmma<64, true><<<gemmBlocks1, 256, 0, stream>>>(
        msg, deg, x, W1l, b1, W1r, h);

    // ---- layer 2 (degrees unchanged; reuse) ----
    sage_zero_f32<<<(nzero2 + 255) / 256, 256, 0, stream>>>(msg, nzero2);
    sage_scatter<<<scatBlocks, 256, 0, stream>>>(h, src, dst, msg, nullptr);
    sage_layer_wmma<16, false><<<gemmBlocks2, 256, 0, stream>>>(
        msg, deg, h, W2l, b2, W2r, out);
}
